// LSTM_30_42537356100405
// MI455X (gfx1250) — matrix operand order repaired and checked by reading, not compiled, not run on hardware
//
#include <hip/hip_runtime.h>
#include <hip/hip_bf16.h>
#include <stdint.h>

// ---------------------------------------------------------------------------
// 2-layer LSTM, 30 steps, B=H=512, per-step weights (reference reuses x[28]
// for step 29). Per (step,layer): one bf16-WMMA GEMM (K folded to 1024 over
// [X | h_prev] @ [W_ih ; W_hh]^T) + one elementwise cell kernel.
// Weight traffic (480 MB) sets the floor (~21us @ 23.3 TB/s); bf16 WMMA with
// f32 accumulation keeps compute off the critical path.
// ---------------------------------------------------------------------------

typedef __attribute__((ext_vector_type(16))) __bf16 v16bf;
typedef __attribute__((ext_vector_type(8)))  __bf16 v8bf;
typedef __attribute__((ext_vector_type(2)))  __bf16 v2bf;
typedef __attribute__((ext_vector_type(8)))  float  v8f;

#define HDIM  512
#define BDIM  512
#define FOURH 2048
#define BM    128
#define BN    64
#define BK    32
#define LDA   40   // padded LDS row stride (bf16 elems) = 80B, 16B-aligned
#define LDB   40

// fp32 -> packed bf16x2. Prefer the native packed convert; otherwise plain
// casts (native v_cvt on targets with hardware bf16, which gfx1250 has).
__device__ __forceinline__ uint32_t pack2(float a, float b) {
#if __has_builtin(__builtin_amdgcn_cvt_pk_bf16_f32)
  union { v2bf v; uint32_t u; } cv;
  cv.v = __builtin_amdgcn_cvt_pk_bf16_f32(a, b);
  return cv.u;
#else
  union { __bf16 h[2]; uint32_t u; } cv;
  cv.h[0] = (__bf16)a;
  cv.h[1] = (__bf16)b;
  return cv.u;
#endif
}

// gates[B,4H] = A0[B,H] @ Wih^T + A1[B,H] @ Whh^T   (single K=1024 reduction)
__global__ __launch_bounds__(256)
void lstm_gates_gemm(const float* __restrict__ A0,   // layer input  [B,H]
                     const float* __restrict__ A1,   // h_prev       [B,H]
                     const float* __restrict__ Wih,  // [4H,H] row-major
                     const float* __restrict__ Whh,  // [4H,H] row-major
                     float* __restrict__ gates) {    // [B,4H]
  __shared__ __bf16 As[2][BM * LDA];
  __shared__ __bf16 Bs[2][BN * LDB];

  const int tid   = threadIdx.x;
  const int lane  = tid & 31;
  const int wid   = tid >> 5;
  const int waveM = (wid & 3) * 32;      // 4 waves x 32 rows = 128
  const int waveN = (wid >> 2) * 32;     // 2 waves x 32 cols = 64
  const int m0 = blockIdx.y * BM;
  const int n0 = blockIdx.x * BN;

  // staging assignments: A = 2 threads/row x 16 floats, B = 4 threads/row x 8
  const int ar = tid >> 1, ak = (tid & 1) * 16;
  const int br = tid >> 2, bk = (tid & 3) * 8;

  v8f acc[2][2];
#pragma unroll
  for (int mt = 0; mt < 2; ++mt)
#pragma unroll
    for (int nt = 0; nt < 2; ++nt)
      acc[mt][nt] = (v8f){0.f, 0.f, 0.f, 0.f, 0.f, 0.f, 0.f, 0.f};

  auto stage = [&](int kt, int buf) {
    const int kb = kt * BK;
    const float* Asrc = (kb < HDIM) ? (A0 + kb)  : (A1 + (kb - HDIM));
    const float* Wsrc = (kb < HDIM) ? (Wih + kb) : (Whh + (kb - HDIM));
    // ---- A tile: [BM][BK] fp32 -> bf16, row-major (matches X memory order)
    const float4* ap = (const float4*)(Asrc + (size_t)(m0 + ar) * HDIM + ak);
    float4 a0 = ap[0], a1 = ap[1], a2 = ap[2], a3 = ap[3];
    uint4 pa0 = { pack2(a0.x, a0.y), pack2(a0.z, a0.w),
                  pack2(a1.x, a1.y), pack2(a1.z, a1.w) };
    uint4 pa1 = { pack2(a2.x, a2.y), pack2(a2.z, a2.w),
                  pack2(a3.x, a3.y), pack2(a3.z, a3.w) };
    *(uint4*)&As[buf][ar * LDA + ak]     = pa0;
    *(uint4*)&As[buf][ar * LDA + ak + 8] = pa1;
    // ---- B tile: [BN][BK], per output-column contiguous K (W's native order)
    const float4* bp = (const float4*)(Wsrc + (size_t)(n0 + br) * HDIM + bk);
    float4 b0 = bp[0], b1 = bp[1];
    uint4 pb = { pack2(b0.x, b0.y), pack2(b0.z, b0.w),
                 pack2(b1.x, b1.y), pack2(b1.z, b1.w) };
    *(uint4*)&Bs[buf][br * LDB + bk] = pb;
  };

  stage(0, 0);
  __syncthreads();

  const int arow   = lane & 15;
  const int klaneA = (lane < 16) ? 0 : 8;    // A: K {0-7,16-23} / {8-15,24-31}
  const int klaneB = (lane < 16) ? 0 : 8;    // B: K {0-7,16-23} / {8-15,24-31}

  for (int kt = 0; kt < (2 * HDIM) / BK; ++kt) {
    const int buf = kt & 1;
    if (kt + 1 < (2 * HDIM) / BK) stage(kt + 1, buf ^ 1);

    v16bf aF[2], bF[2];
#pragma unroll
    for (int mt = 0; mt < 2; ++mt) {
      const int r = waveM + mt * 16 + arow;
      union { v16bf v; v8bf h[2]; } u;
      u.h[0] = *(const v8bf*)&As[buf][r * LDA + klaneA];
      u.h[1] = *(const v8bf*)&As[buf][r * LDA + klaneA + 16];
      aF[mt] = u.v;
    }
#pragma unroll
    for (int nt = 0; nt < 2; ++nt) {
      const int j = waveN + nt * 16 + arow;
      union { v16bf v; v8bf h[2]; } u;
      u.h[0] = *(const v8bf*)&Bs[buf][j * LDB + klaneB];
      u.h[1] = *(const v8bf*)&Bs[buf][j * LDB + klaneB + 16];
      bF[nt] = u.v;
    }
#pragma unroll
    for (int mt = 0; mt < 2; ++mt)
#pragma unroll
      for (int nt = 0; nt < 2; ++nt)
        acc[mt][nt] = __builtin_amdgcn_wmma_f32_16x16x32_bf16(
            false, aF[mt], false, bF[nt], (short)0, acc[mt][nt], false, false);

    __syncthreads();
  }

  // C/D layout: VGPR vr -> M = vr + (lane<16 ? 0 : 8), N = lane & 15
  const int mhalf = (lane < 16) ? 0 : 8;
  const int ncol  = lane & 15;
#pragma unroll
  for (int mt = 0; mt < 2; ++mt)
#pragma unroll
    for (int nt = 0; nt < 2; ++nt) {
      const int gcol = n0 + waveN + nt * 16 + ncol;
#pragma unroll
      for (int vr = 0; vr < 8; ++vr) {
        const int grow = m0 + waveM + mt * 16 + mhalf + vr;
        gates[(size_t)grow * FOURH + gcol] = acc[mt][nt][vr];
      }
    }
}

__device__ __forceinline__ float sigm_(float x) { return 1.f / (1.f + __expf(-x)); }
__device__ __forceinline__ float tanh_(float x) { return 1.f - 2.f / (1.f + __expf(2.f * x)); }

__global__ __launch_bounds__(256)
void lstm_cell(const float* __restrict__ gates,  // [B,4H]
               const float* __restrict__ bih,    // [4H]
               const float* __restrict__ bhh,    // [4H]
               float* __restrict__ c,            // [B,H] in/out
               float* __restrict__ h,            // [B,H] out
               float* __restrict__ y) {          // [B,H] out or null
  const int idx = blockIdx.x * blockDim.x + threadIdx.x;
  if (idx >= BDIM * HDIM) return;
  const int b = idx >> 9;
  const int n = idx & (HDIM - 1);
  const float* g = gates + (size_t)b * FOURH;
  const float gi = g[n]            + bih[n]            + bhh[n];
  const float gf = g[n + HDIM]     + bih[n + HDIM]     + bhh[n + HDIM];
  const float gg = g[n + 2 * HDIM] + bih[n + 2 * HDIM] + bhh[n + 2 * HDIM];
  const float go = g[n + 3 * HDIM] + bih[n + 3 * HDIM] + bhh[n + 3 * HDIM];
  const float i = sigm_(gi), f = sigm_(gf), t = tanh_(gg), o = sigm_(go);
  const float cn = f * c[idx] + i * t;
  c[idx] = cn;
  const float hn = o * tanh_(cn);
  h[idx] = hn;
  if (y) y[idx] = hn;
}

__global__ void zero_kernel(float* p, int n) {
  const int i = blockIdx.x * blockDim.x + threadIdx.x;
  if (i < n) p[i] = 0.f;
}

extern "C" void kernel_launch(void* const* d_in, const int* in_sizes, int n_in,
                              void* d_out, int out_size, void* d_ws, size_t ws_size,
                              hipStream_t stream) {
  const float* x   = (const float*)d_in[0];  // [30,B,H]
  const float* Wih = (const float*)d_in[1];  // [30,2,4H,H]
  const float* Whh = (const float*)d_in[2];  // [30,2,4H,H]
  const float* bih = (const float*)d_in[3];  // [30,2,4H]
  const float* bhh = (const float*)d_in[4];  // [30,2,4H]
  float* y = (float*)d_out;                  // ys[30,B,H] ++ h[2,B,H] ++ c[2,B,H]

  const size_t BH = (size_t)BDIM * HDIM;
  float* gates = (float*)d_ws;               // B*4H floats
  float* hbuf  = gates + (size_t)BDIM * FOURH;
  float* cbuf  = hbuf + 2 * BH;              // total ws: 8 MB

  {
    const int n = (int)(4 * BH);
    zero_kernel<<<(n + 255) / 256, 256, 0, stream>>>(hbuf, n);  // h0=c0=0
  }

  const dim3 gGemm(FOURH / BN, BDIM / BM);   // 32 x 4 = 128 blocks
  const int cellBlocks = (int)((BH + 255) / 256);

  for (int t = 0; t < 30; ++t) {
    const int xi = (t < 29) ? t : 28;        // reference reuses x[28] at t=29
    const float* xt = x + (size_t)xi * BH;
    for (int l = 0; l < 2; ++l) {
      const float* a0 = (l == 0) ? xt : hbuf;             // layer input
      const float* a1 = hbuf + (size_t)l * BH;            // this layer's h_prev
      const size_t wofs = ((size_t)t * 2 + l) * (size_t)FOURH * HDIM;
      const size_t bofs = ((size_t)t * 2 + l) * FOURH;
      lstm_gates_gemm<<<gGemm, 256, 0, stream>>>(
          a0, a1, Wih + wofs, Whh + wofs, gates);
      lstm_cell<<<cellBlocks, 256, 0, stream>>>(
          gates, bih + bofs, bhh + bofs,
          cbuf + (size_t)l * BH, hbuf + (size_t)l * BH,
          (l == 1) ? (y + (size_t)t * BH) : nullptr);
    }
  }

  hipMemcpyAsync(y + 30 * BH, hbuf, 2 * BH * sizeof(float),
                 hipMemcpyDeviceToDevice, stream);
  hipMemcpyAsync(y + 32 * BH, cbuf, 2 * BH * sizeof(float),
                 hipMemcpyDeviceToDevice, stream);
}
